// GroupedQueryAttention_62612033241430
// MI455X (gfx1250) — matrix operand order repaired and checked by reading, not compiled, not run on hardware
//
#include <hip/hip_runtime.h>

// ---------------------------------------------------------------------------
// GQA forward for MI455X (gfx1250, wave32, WMMA).
// Pipeline: cvt -> QKV GEMM (bf16 WMMA, double-buffered) -> RoPE/relayout ->
// flash attention (bf16 WMMA, fp32 softmax) -> output GEMM -> fp32 out.
// ---------------------------------------------------------------------------

typedef __attribute__((ext_vector_type(16))) __bf16 bf16x16;
typedef __attribute__((ext_vector_type(8)))  float  v8f;

union Frag16 { bf16x16 v; uint4 q[2]; };

constexpr int B_   = 2;
constexpr int T_   = 2048;
constexpr int D_   = 2048;
constexpr int H_   = 16;
constexpr int HKV_ = 4;
constexpr int HD_  = 128;
constexpr int KVD_ = HKV_ * HD_;   // 512
constexpr int M_   = B_ * T_;      // 4096 rows in all GEMMs

__device__ __forceinline__ __bf16 f2bf(float f) {
  // round-to-nearest-even fp32 -> bf16
  unsigned u = __float_as_uint(f);
  unsigned r = u + 0x7FFFu + ((u >> 16) & 1u);
  unsigned short s = (unsigned short)(r >> 16);
  return __builtin_bit_cast(__bf16, s);
}

__global__ void cvt_f32_bf16(const float* __restrict__ src,
                             __bf16* __restrict__ dst, int n) {
  int i = blockIdx.x * blockDim.x + threadIdx.x;
  if (i < n) dst[i] = f2bf(src[i]);
}

// ---------------------------------------------------------------------------
// GEMM: C[m,n] = sum_k A[m,k] * W[n,k]   (A: Mm x Kk bf16, W: Nn x Kk bf16)
// One wave computes a 32x64 tile: 2 A-fragments x 4 B-fragments -> 8 WMMAs
// per 32-wide k-step, with register double-buffering so the next k-step's 12
// b128 loads are in flight while the current 8 WMMAs execute.
// Fragment layouts used here (wave32; contiguous k, not the ISA table order):
//   A 16x32 bf16:  lane<16 -> row=l16, K = k0..k0+15 (contiguous)
//                  lane>=16 -> row=l16, K = k0+16..k0+31
//   B 32x16 bf16:  lane<16 -> col=l16, K = k0..k0+15 (contiguous)
//                  lane>=16 -> col=l16, K = k0+16..k0+31
// Both operands are K-major in memory -> two 16B loads per lane each.
// ---------------------------------------------------------------------------
struct GemmBufs { Frag16 a0, a1, b[4]; };

__device__ __forceinline__ void gemm_load(GemmBufs& g,
                                          const __bf16* __restrict__ arow0,
                                          const __bf16* __restrict__ arow1,
                                          const __bf16* const brow[4],
                                          int k0) {
  g.a0.q[0] = *(const uint4*)(arow0 + k0);
  g.a0.q[1] = *(const uint4*)(arow0 + k0 + 8);
  g.a1.q[0] = *(const uint4*)(arow1 + k0);
  g.a1.q[1] = *(const uint4*)(arow1 + k0 + 8);
#pragma unroll
  for (int j = 0; j < 4; ++j) {
    g.b[j].q[0] = *(const uint4*)(brow[j] + k0);
    g.b[j].q[1] = *(const uint4*)(brow[j] + k0 + 8);
  }
}

__device__ __forceinline__ void gemm_mma(const GemmBufs& g, v8f acc[2][4]) {
#pragma unroll
  for (int j = 0; j < 4; ++j) {
    acc[0][j] = __builtin_amdgcn_wmma_f32_16x16x32_bf16(
        false, g.a0.v, false, g.b[j].v, (short)0, acc[0][j], false, false);
    acc[1][j] = __builtin_amdgcn_wmma_f32_16x16x32_bf16(
        false, g.a1.v, false, g.b[j].v, (short)0, acc[1][j], false, false);
  }
}

__global__ __launch_bounds__(256) void gemm_bf16_nt(
    const __bf16* __restrict__ A, const __bf16* __restrict__ W,
    float* __restrict__ C, int Mm, int Nn, int Kk) {
  const int wave = threadIdx.x >> 5;
  const int lane = threadIdx.x & 31;
  const int half = lane >> 4;
  const int l16  = lane & 15;
  const int ntiles = Nn >> 6;                 // 64-wide N tiles
  const int tile = blockIdx.x * 8 + wave;
  if (tile >= (Mm >> 5) * ntiles) return;     // wave-uniform guard
  const int tm = tile / ntiles;               // 32-row M tiles
  const int tn = tile % ntiles;

  const __bf16* arow0 = A + (size_t)(tm * 32 + l16) * Kk + half * 16;
  const __bf16* arow1 = A + (size_t)(tm * 32 + 16 + l16) * Kk + half * 16;
  const __bf16* brow[4];
#pragma unroll
  for (int j = 0; j < 4; ++j)
    brow[j] = W + (size_t)(tn * 64 + j * 16 + l16) * Kk + half * 16;

  v8f acc[2][4] = {};
  GemmBufs bufA, bufB;
  gemm_load(bufA, arow0, arow1, brow, 0);
  int k0 = 0;
  for (; k0 + 64 < Kk; k0 += 64) {  // Kk is a multiple of 64 (=2048)
    // prefetch 8 k-steps ahead (global_prefetch_b8)
    __builtin_prefetch(arow0 + k0 + 512, 0, 1);
    __builtin_prefetch(arow1 + k0 + 512, 0, 1);
#pragma unroll
    for (int j = 0; j < 4; ++j) __builtin_prefetch(brow[j] + k0 + 512, 0, 1);
    gemm_load(bufB, arow0, arow1, brow, k0 + 32);  // in flight during...
    gemm_mma(bufA, acc);                           // ...these 8 WMMAs
    gemm_load(bufA, arow0, arow1, brow, k0 + 64);
    gemm_mma(bufB, acc);
  }
  gemm_load(bufB, arow0, arow1, brow, k0 + 32);
  gemm_mma(bufA, acc);
  gemm_mma(bufB, acc);

  // C/D layout: vgpr r, lane<16 -> (m=r, n=l16); lane>=16 -> (m=r+8, n=l16)
#pragma unroll
  for (int i = 0; i < 2; ++i)
#pragma unroll
    for (int j = 0; j < 4; ++j)
#pragma unroll
      for (int r = 0; r < 8; ++r)
        C[(size_t)(tm * 32 + i * 16 + r + half * 8) * Nn + tn * 64 + j * 16 +
          l16] = acc[i][j][r];
}

// ---------------------------------------------------------------------------
// RoPE on q: qf (B*T, D) fp32 -> qb [b][h][t][hd] bf16 (head-major)
// ---------------------------------------------------------------------------
__global__ void rope_q_kernel(const float* __restrict__ qf,
                              const float* __restrict__ cosT,
                              const float* __restrict__ sinT,
                              __bf16* __restrict__ qb) {
  unsigned idx = blockIdx.x * blockDim.x + threadIdx.x;  // B*T*H*64
  int i = idx & 63;
  unsigned rest = idx >> 6;
  int h = rest & (H_ - 1); rest >>= 4;
  int t = rest & (T_ - 1);
  int b = rest >> 11;
  const float* src = qf + (size_t)(b * T_ + t) * D_ + h * HD_;
  float x1 = src[i], x2 = src[i + 64];
  float c1 = cosT[t * HD_ + i],      s1 = sinT[t * HD_ + i];
  float c2 = cosT[t * HD_ + i + 64], s2 = sinT[t * HD_ + i + 64];
  __bf16* dst = qb + (size_t)((b * H_ + h) * T_ + t) * HD_;
  dst[i]      = f2bf(x1 * c1 - x2 * s1);
  dst[i + 64] = f2bf(x2 * c2 + x1 * s2);
}

__global__ void rope_k_kernel(const float* __restrict__ kf,
                              const float* __restrict__ cosT,
                              const float* __restrict__ sinT,
                              __bf16* __restrict__ kb) {
  unsigned idx = blockIdx.x * blockDim.x + threadIdx.x;  // B*T*HKV*64
  int i = idx & 63;
  unsigned rest = idx >> 6;
  int h = rest & (HKV_ - 1); rest >>= 2;
  int t = rest & (T_ - 1);
  int b = rest >> 11;
  const float* src = kf + (size_t)(b * T_ + t) * KVD_ + h * HD_;
  float x1 = src[i], x2 = src[i + 64];
  float c1 = cosT[t * HD_ + i],      s1 = sinT[t * HD_ + i];
  float c2 = cosT[t * HD_ + i + 64], s2 = sinT[t * HD_ + i + 64];
  __bf16* dst = kb + (size_t)((b * HKV_ + h) * T_ + t) * HD_;
  dst[i]      = f2bf(x1 * c1 - x2 * s1);
  dst[i + 64] = f2bf(x2 * c2 + x1 * s2);
}

// V: vf (B*T, KVD) fp32 -> vt [b][kvh][hd][t] bf16 (transposed so the PV
// B-fragment reads K(=kv) contiguously per lane).
__global__ void v_relayout_kernel(const float* __restrict__ vf,
                                  __bf16* __restrict__ vt) {
  unsigned idx = blockIdx.x * blockDim.x + threadIdx.x;  // B*HKV*HD*T
  int t = idx & (T_ - 1);
  unsigned rest = idx >> 11;
  int hd = rest & (HD_ - 1); rest >>= 7;
  int kvh = rest & (HKV_ - 1);
  int b = rest >> 2;
  float val = vf[(size_t)(b * T_ + t) * KVD_ + kvh * HD_ + hd];
  vt[((size_t)(b * HKV_ + kvh) * HD_ + hd) * T_ + t] = f2bf(val);
}

__device__ __forceinline__ float rowmax16(float v) {
  v = fmaxf(v, __shfl_xor(v, 1, 32));
  v = fmaxf(v, __shfl_xor(v, 2, 32));
  v = fmaxf(v, __shfl_xor(v, 4, 32));
  v = fmaxf(v, __shfl_xor(v, 8, 32));
  return v;
}
__device__ __forceinline__ float rowsum16(float v) {
  v += __shfl_xor(v, 1, 32);
  v += __shfl_xor(v, 2, 32);
  v += __shfl_xor(v, 4, 32);
  v += __shfl_xor(v, 8, 32);
  return v;
}

// ---------------------------------------------------------------------------
// Flash attention: one wave per (b, h, 16-row q tile); KV streamed in chunks
// of 32 with online softmax. S = Q K^T via 8 WMMAs/chunk, O += P V via 8
// WMMAs/chunk. P is moved C-layout -> A-layout through a wave-private LDS
// tile (wave-local ordering only: s_wait_dscnt, no block barrier, so the
// causal (divergent-across-waves) trip counts are safe).
// ---------------------------------------------------------------------------
__global__ __launch_bounds__(128) void flash_attn_kernel(
    const __bf16* __restrict__ qb, const __bf16* __restrict__ kb,
    const __bf16* __restrict__ vt, __bf16* __restrict__ attnb) {
  __shared__ __align__(16) __bf16 ldsP[4][16 * 32];
  const int wave = threadIdx.x >> 5;
  const int lane = threadIdx.x & 31;
  const int half = lane >> 4;
  const int l16  = lane & 15;
  const int tile = blockIdx.x * 4 + wave;      // B*H*(T/16) = 4096 tiles
  const int qt = tile & (T_ / 16 - 1);
  int rest = tile >> 7;
  const int h = rest & (H_ - 1);
  const int b = rest >> 4;
  const int kvh = h >> 2;                      // G = H/HKV = 4
  const int qbase = qt * 16;
  __bf16* myP = ldsP[wave];

  // Preload the 4 Q A-fragments (HD=128 -> 4 K-steps of 32), kept resident.
  const __bf16* qrow = qb + ((size_t)((b * H_ + h) * T_) + qbase + l16) * HD_;
  Frag16 qfr[4];
#pragma unroll
  for (int kk = 0; kk < 4; ++kk) {
    const __bf16* p = qrow + kk * 32 + half * 16;
    qfr[kk].q[0] = *(const uint4*)(p);
    qfr[kk].q[1] = *(const uint4*)(p + 8);
  }

  const __bf16* kbp = kb + (size_t)(b * HKV_ + kvh) * T_ * HD_;
  const __bf16* vbp = vt + (size_t)(b * HKV_ + kvh) * HD_ * T_;

  v8f O[8] = {};
  float mrow[8], lrow[8];
#pragma unroll
  for (int r = 0; r < 8; ++r) { mrow[r] = -3.0e38f; lrow[r] = 0.0f; }

  const float scale = 0.08838834764831845f;    // 1/sqrt(128)
  const int nchunk = (qbase + 47) >> 5;        // causal: kv <= qbase+15
  for (int kc = 0; kc < nchunk; ++kc) {
    const int kv0 = kc * 32;
    const bool pre = (kc + 1) < nchunk;        // wave-uniform
    if (pre) {                                 // prefetch next chunk's K rows
      const __bf16* kpre = kbp + (size_t)(kv0 + 32 + l16) * HD_;
      __builtin_prefetch(kpre, 0, 1);
      __builtin_prefetch(kpre + 16 * HD_, 0, 1);
    }
    v8f S0 = {}, S1 = {};
#pragma unroll
    for (int kk = 0; kk < 4; ++kk) {
      const __bf16* kr0 = kbp + (size_t)(kv0 + l16) * HD_ + kk * 32 + half * 16;
      Frag16 kf0;
      kf0.q[0] = *(const uint4*)(kr0);
      kf0.q[1] = *(const uint4*)(kr0 + 8);
      S0 = __builtin_amdgcn_wmma_f32_16x16x32_bf16(
          false, qfr[kk].v, false, kf0.v, (short)0, S0, false, false);
      const __bf16* kr1 =
          kbp + (size_t)(kv0 + 16 + l16) * HD_ + kk * 32 + half * 16;
      Frag16 kf1;
      kf1.q[0] = *(const uint4*)(kr1);
      kf1.q[1] = *(const uint4*)(kr1 + 8);
      S1 = __builtin_amdgcn_wmma_f32_16x16x32_bf16(
          false, qfr[kk].v, false, kf1.v, (short)0, S1, false, false);
    }
    // scale + causal mask (finite sentinel: no inf-inf NaN paths)
    const int n0 = kv0 + l16;
    const int n1 = kv0 + 16 + l16;
#pragma unroll
    for (int r = 0; r < 8; ++r) {
      const int qg = qbase + r + half * 8;
      float s0 = S0[r] * scale;
      float s1 = S1[r] * scale;
      if (n0 > qg) s0 = -3.0e38f;
      if (n1 > qg) s1 = -3.0e38f;
      S0[r] = s0; S1[r] = s1;
    }
    // online softmax update
    float alpha[8];
#pragma unroll
    for (int r = 0; r < 8; ++r) {
      float mnew = fmaxf(mrow[r], rowmax16(fmaxf(S0[r], S1[r])));
      alpha[r] = __expf(mrow[r] - mnew);
      mrow[r] = mnew;
      float p0 = __expf(S0[r] - mnew);
      float p1 = __expf(S1[r] - mnew);
      S0[r] = p0; S1[r] = p1;
      lrow[r] = lrow[r] * alpha[r] + rowsum16(p0 + p1);
    }
#pragma unroll
    for (int d = 0; d < 8; ++d)
#pragma unroll
      for (int r = 0; r < 8; ++r) O[d][r] *= alpha[r];

    // P: C-layout -> LDS (row-major 16x32) -> A-fragment
#pragma unroll
    for (int r = 0; r < 8; ++r) {
      myP[(r + half * 8) * 32 + l16]      = f2bf(S0[r]);
      myP[(r + half * 8) * 32 + 16 + l16] = f2bf(S1[r]);
    }
    asm volatile("s_wait_dscnt 0x0" ::: "memory");  // wave-local store->load
    Frag16 pa;
    pa.q[0] = *(const uint4*)(myP + l16 * 32 + half * 16);
    pa.q[1] = *(const uint4*)(myP + l16 * 32 + half * 16 + 8);

    // O += P @ V  (B fragment from transposed V: K=kv contiguous per lane)
#pragma unroll
    for (int d = 0; d < 8; ++d) {
      const __bf16* vr = vbp + (size_t)(d * 16 + l16) * T_ + kv0 + half * 16;
      Frag16 vfr;
      vfr.q[0] = *(const uint4*)(vr);
      vfr.q[1] = *(const uint4*)(vr + 8);
      if (pre) __builtin_prefetch(vr + 32, 0, 1);  // next chunk's V line
      O[d] = __builtin_amdgcn_wmma_f32_16x16x32_bf16(
          false, pa.v, false, vfr.v, (short)0, O[d], false, false);
    }
  }

  // normalize + emit bf16 attn in (B*T, D) layout for the output projection
  float inv[8];
#pragma unroll
  for (int r = 0; r < 8; ++r) inv[r] = 1.0f / lrow[r];
#pragma unroll
  for (int d = 0; d < 8; ++d)
#pragma unroll
    for (int r = 0; r < 8; ++r) {
      size_t row = (size_t)(b * T_ + qbase + r + half * 8);
      attnb[row * D_ + h * HD_ + d * 16 + l16] = f2bf(O[d][r] * inv[r]);
    }
}

// ---------------------------------------------------------------------------
extern "C" void kernel_launch(void* const* d_in, const int* in_sizes, int n_in,
                              void* d_out, int out_size, void* d_ws,
                              size_t ws_size, hipStream_t stream) {
  const float* x    = (const float*)d_in[0];
  const float* cosT = (const float*)d_in[1];
  const float* sinT = (const float*)d_in[2];
  const float* Wq   = (const float*)d_in[3];
  const float* Wk   = (const float*)d_in[4];
  const float* Wv   = (const float*)d_in[5];
  const float* Wo   = (const float*)d_in[6];
  float* out = (float*)d_out;

  char* ws = (char*)d_ws;
  size_t off = 0;
  auto carve = [&](size_t bytes) -> void* {
    void* p = ws + off;
    off += (bytes + 255) & ~(size_t)255;
    return p;
  };
  __bf16* xb    = (__bf16*)carve((size_t)M_ * D_ * 2);
  __bf16* Wqb   = (__bf16*)carve((size_t)D_ * D_ * 2);
  __bf16* Wkb   = (__bf16*)carve((size_t)KVD_ * D_ * 2);
  __bf16* Wvb   = (__bf16*)carve((size_t)KVD_ * D_ * 2);
  __bf16* Wob   = (__bf16*)carve((size_t)D_ * D_ * 2);
  float*  qf    = (float*) carve((size_t)M_ * D_ * 4);
  float*  kf    = (float*) carve((size_t)M_ * KVD_ * 4);
  float*  vf    = (float*) carve((size_t)M_ * KVD_ * 4);
  __bf16* qbb   = (__bf16*)carve((size_t)M_ * D_ * 2);
  __bf16* kbb   = (__bf16*)carve((size_t)M_ * KVD_ * 2);
  __bf16* vtb   = (__bf16*)carve((size_t)M_ * KVD_ * 2);
  __bf16* attnb = (__bf16*)carve((size_t)M_ * D_ * 2);

  // 1) fp32 -> bf16 conversions
  cvt_f32_bf16<<<(M_ * D_) / 256, 256, 0, stream>>>(x, xb, M_ * D_);
  cvt_f32_bf16<<<(D_ * D_) / 256, 256, 0, stream>>>(Wq, Wqb, D_ * D_);
  cvt_f32_bf16<<<(KVD_ * D_) / 256, 256, 0, stream>>>(Wk, Wkb, KVD_ * D_);
  cvt_f32_bf16<<<(KVD_ * D_) / 256, 256, 0, stream>>>(Wv, Wvb, KVD_ * D_);
  cvt_f32_bf16<<<(D_ * D_) / 256, 256, 0, stream>>>(Wo, Wob, D_ * D_);

  // 2) QKV projections (WMMA GEMM): tiles = (M/32)*(N/64), 8 waves/block
  gemm_bf16_nt<<<(M_ / 32) * (D_ / 64) / 8, 256, 0, stream>>>(
      xb, Wqb, qf, M_, D_, D_);
  gemm_bf16_nt<<<(M_ / 32) * (KVD_ / 64) / 8, 256, 0, stream>>>(
      xb, Wkb, kf, M_, KVD_, D_);
  gemm_bf16_nt<<<(M_ / 32) * (KVD_ / 64) / 8, 256, 0, stream>>>(
      xb, Wvb, vf, M_, KVD_, D_);

  // 3) RoPE + relayout
  rope_q_kernel<<<(B_ * T_ * H_ * 64) / 256, 256, 0, stream>>>(qf, cosT, sinT,
                                                               qbb);
  rope_k_kernel<<<(B_ * T_ * HKV_ * 64) / 256, 256, 0, stream>>>(kf, cosT,
                                                                 sinT, kbb);
  v_relayout_kernel<<<(B_ * HKV_ * HD_ * T_) / 256, 256, 0, stream>>>(vf, vtb);

  // 4) flash attention: B*H*(T/16) = 4096 wave-tiles, 4 waves/block
  flash_attn_kernel<<<(B_ * H_ * (T_ / 16)) / 4, 128, 0, stream>>>(qbb, kbb,
                                                                   vtb, attnb);

  // 5) output projection -> fp32 d_out
  gemm_bf16_nt<<<(M_ / 32) * (D_ / 64) / 8, 256, 0, stream>>>(
      attnb, Wob, out, M_, D_, D_);
}
